// Modelv6_28114855919775
// MI455X (gfx1250) — matrix operand order repaired and checked by reading, not compiled, not run on hardware
//
#include <hip/hip_runtime.h>
#include <stdint.h>

#define N_OER     100000
#define N_CONCEPT 20000
#define N_CLASS   1000
#define F_ENT     64
#define HID       128
#define E_SR      200000
#define E_EP      400000
#define E_COV     400000
#define E_BEL     100000
#define NEG_SLOPE 0.2f
#define ROWS_PER_BLOCK 64

typedef __attribute__((ext_vector_type(16))) _Float16 v16h;
typedef __attribute__((ext_vector_type(8)))  float    v8f;
typedef __attribute__((ext_vector_type(4)))  int      v4i;

typedef __attribute__((address_space(1))) v4i* gv4i_p;   // global int4*
typedef __attribute__((address_space(3))) v4i* lv4i_p;   // LDS int4*

#ifndef __has_builtin
#define __has_builtin(x) 0
#endif
#if __has_builtin(__builtin_amdgcn_global_load_async_to_lds_b128)
#define HAVE_ASYNC_LDS 1
#else
#define HAVE_ASYNC_LDS 0
#endif

__device__ __forceinline__ void wait_asynccnt0() {
#if __has_builtin(__builtin_amdgcn_s_wait_asynccnt)
  __builtin_amdgcn_s_wait_asynccnt(0);
#else
  asm volatile("s_wait_asynccnt 0x0" ::: "memory");
#endif
}

// ---------------------------------------------------------------------------
// GEMM: Y[M x 128] = X[M x K] @ W[K x 128] (+ bias), f16 WMMA, f32 accumulate.
// Block = 256 threads = 8 waves; block owns a 64-row stripe of Y; wave w owns
// the 16-column tile at n = 16*w and 4 stacked M-tiles (4 accumulators), so
// each B fragment is reused by 4 v_wmma. A stripe (64 x K f32, contiguous in
// memory) is staged to LDS via async global->LDS b128 when available.
// A/B lane k order: contiguous k = 16*(lane/16) + s (wave32).
// ---------------------------------------------------------------------------
__global__ void gemm_wmma_f16(const float* __restrict__ X,
                              const float* __restrict__ W,
                              const float* __restrict__ bias,
                              float* __restrict__ Y, int M, int K) {
  __shared__ float Ash[ROWS_PER_BLOCK * 128];   // K <= 128 -> 32 KB
  const int tid  = threadIdx.x;
  const int wave = tid >> 5;
  const int lane = tid & 31;
  const int row0 = blockIdx.x * ROWS_PER_BLOCK;
  const int nElems = ROWS_PER_BLOCK * K;

#if HAVE_ASYNC_LDS
  if (row0 + ROWS_PER_BLOCK <= M) {
    // whole stripe in-bounds: rows are contiguous -> flat async b128 copy
    const float* gbase = X + (size_t)row0 * K;
    for (int idx4 = tid * 4; idx4 < nElems; idx4 += 256 * 4) {
      __builtin_amdgcn_global_load_async_to_lds_b128(
          (gv4i_p)(gbase + idx4), (lv4i_p)(Ash + idx4), 0, 0);
    }
    wait_asynccnt0();
  } else
#endif
  {
    for (int idx = tid; idx < nElems; idx += 256) {
      int r = idx / K, k = idx - r * K;
      int gr = row0 + r;
      Ash[idx] = (gr < M) ? X[(size_t)gr * K + k] : 0.0f;
    }
  }
  __syncthreads();

  const int nbase = wave * 16;
  const int m  = lane & 15;                  // A row within tile
  const int ka = (lane < 16) ? 0 : 16;       // A: K=0..15 vs 16..31
  const int n  = lane & 15;                  // B/C column
  const int kb = (lane < 16) ? 0 : 16;       // B: K=0..15 vs 16..31

  v8f c0 = {}, c1 = {}, c2 = {}, c3 = {};
  for (int k0 = 0; k0 < K; k0 += 32) {
    v16h b;
    const float* Wp = &W[(size_t)(k0 + kb) * HID + nbase + n];
#pragma unroll
    for (int j = 0; j < 16; ++j) b[j] = (_Float16)Wp[(size_t)j * HID];

#pragma unroll
    for (int t = 0; t < 4; ++t) {
      v16h a;
      const float* Ar = &Ash[(t * 16 + m) * K + k0];
#pragma unroll
      for (int j = 0; j < 8; ++j) {
        a[j]     = (_Float16)Ar[ka + j];
        a[8 + j] = (_Float16)Ar[ka + 8 + j];
      }
      v8f& cc = (t == 0) ? c0 : (t == 1) ? c1 : (t == 2) ? c2 : c3;
      // (neg_a, A, neg_b, B, c_mod, C, reuse_a, reuse_b)
      cc = __builtin_amdgcn_wmma_f32_16x16x32_f16(false, a, false, b, (short)0,
                                                  cc, false, false);
    }
  }

  // C/D layout: VGPR j -> row j (lanes 0-15) or 8+j (lanes 16-31), col = lane&15
  const int   col = nbase + n;
  const int   rb  = (lane < 16) ? 0 : 8;
  const float bv  = bias ? bias[col] : 0.0f;
  v8f cs[4] = {c0, c1, c2, c3};
#pragma unroll
  for (int t = 0; t < 4; ++t) {
#pragma unroll
    for (int j = 0; j < 8; ++j) {
      int r = row0 + t * 16 + rb + j;
      if (r < M) Y[(size_t)r * HID + col] = cs[t][j] + bv;
    }
  }
}

// ------------------------- small helper kernels ----------------------------
__global__ void fill_f32(float* __restrict__ p, float v, int n) {
  int i = blockIdx.x * blockDim.x + threadIdx.x;
  if (i < n) p[i] = v;
}

__global__ void init_acc_bias(float* __restrict__ acc, const float* __restrict__ b0,
                              const float* __restrict__ b1, int nrows) {
  int i = blockIdx.x * blockDim.x + threadIdx.x;
  if (i < nrows * HID) {
    int c = i & (HID - 1);
    float v = b0[c];
    if (b1) v += b1[c];
    acc[i] = v;
  }
}

__global__ void scale_copy(float* __restrict__ dst, const float* __restrict__ src,
                           float sc, int n) {
  int i = blockIdx.x * blockDim.x + threadIdx.x;
  if (i < n) dst[i] = src[i] * sc;
}

// s[row] = dot(H[row, 0:128], a) ; one wave32 per row
__global__ void rowdot128(const float* __restrict__ H, const float* __restrict__ a,
                          float* __restrict__ s, int N) {
  int row  = blockIdx.x * (blockDim.x >> 5) + (threadIdx.x >> 5);
  int lane = threadIdx.x & 31;
  if (row >= N) return;
  const float* h = H + (size_t)row * HID;
  float acc = h[lane] * a[lane] + h[lane + 32] * a[lane + 32] +
              h[lane + 64] * a[lane + 64] + h[lane + 96] * a[lane + 96];
#pragma unroll
  for (int off = 16; off > 0; off >>= 1) acc += __shfl_down(acc, off, 32);
  if (lane == 0) s[row] = acc;
}

// --------------------------- edge-phase kernels ----------------------------
__device__ __forceinline__ void edge_sd(const int* __restrict__ src,
                                        const int* __restrict__ dst, int E, int e,
                                        int& s, int& d) {
  if (e < E) { s = src[e]; d = dst[e]; } else { s = e - E; d = s; }  // self loops
}

__device__ __forceinline__ void atomicMaxF(float* addr, float val) {
  unsigned int* ua = (unsigned int*)addr;
  unsigned int cur = __float_as_uint(*addr);
  while (__uint_as_float(cur) < val) {
    unsigned int assumed = cur;
    cur = atomicCAS(ua, assumed, __float_as_uint(val));
    if (cur == assumed) break;
  }
}

__global__ void edge_logit_max(const int* __restrict__ src, const int* __restrict__ dst,
                               int E, int nloop, const float* __restrict__ ssrc,
                               const float* __restrict__ sdst, float* __restrict__ logit,
                               float* __restrict__ segmax) {
  int e = blockIdx.x * blockDim.x + threadIdx.x;
  if (e >= E + nloop) return;
  int s, d; edge_sd(src, dst, E, e, s, d);
  float l = ssrc[s] + sdst[d];
  l = (l > 0.0f) ? l : NEG_SLOPE * l;
  logit[e] = l;
  atomicMaxF(&segmax[d], l);
}

__global__ void edge_exp_sum(const int* __restrict__ src, const int* __restrict__ dst,
                             int E, int nloop, const float* __restrict__ logit,
                             const float* __restrict__ segmax, float* __restrict__ ex,
                             float* __restrict__ denom) {
  int e = blockIdx.x * blockDim.x + threadIdx.x;
  if (e >= E + nloop) return;
  int s, d; edge_sd(src, dst, E, e, s, d);
  (void)s;
  float v = __expf(logit[e] - segmax[d]);
  ex[e] = v;
  atomicAdd(&denom[d], v);
}

// one wave32 per edge: float4 gather of hs[src], 4 f32 atomics per lane to acc[dst]
__global__ void edge_scatter(const int* __restrict__ src, const int* __restrict__ dst,
                             int E, int nloop, const float* __restrict__ hs,
                             const float* __restrict__ ex, const float* __restrict__ denom,
                             float* __restrict__ acc) {
  int e    = blockIdx.x * (blockDim.x >> 5) + (threadIdx.x >> 5);
  int lane = threadIdx.x & 31;
  if (e >= E + nloop) return;
  int s, d; edge_sd(src, dst, E, e, s, d);
  __builtin_prefetch(hs + (size_t)s * HID + lane * 4, 0, 0);  // global_prefetch_b8
  float alpha = ex[e] / denom[d];
  const float4 v = ((const float4*)(hs + (size_t)s * HID))[lane];
  float* ad = acc + (size_t)d * HID + lane * 4;
  atomicAdd(ad + 0, v.x * alpha);
  atomicAdd(ad + 1, v.y * alpha);
  atomicAdd(ad + 2, v.z * alpha);
  atomicAdd(ad + 3, v.w * alpha);
}

// ------------------------ final linear classifier --------------------------
// Linear split: out[e] = ns[src] + ms[dst] + b
__global__ void node_scores(const float* __restrict__ x, const float* __restrict__ h,
                            const float* __restrict__ Wc, float* __restrict__ ns,
                            float* __restrict__ ms, int N) {
  int row  = blockIdx.x * (blockDim.x >> 5) + (threadIdx.x >> 5);
  int lane = threadIdx.x & 31;
  if (row >= N) return;
  const float* xr = x + (size_t)row * F_ENT;
  const float* hr = h + (size_t)row * HID;
  float a = 0.f, b = 0.f;
#pragma unroll
  for (int k = 0; k < 2; ++k) {
    int c = lane + 32 * k;
    a += xr[c] * Wc[c];
    b += xr[c] * Wc[192 + c];
  }
#pragma unroll
  for (int k = 0; k < 4; ++k) {
    int c = lane + 32 * k;
    a += hr[c] * Wc[64 + c];
    b += hr[c] * Wc[256 + c];
  }
#pragma unroll
  for (int off = 16; off > 0; off >>= 1) {
    a += __shfl_down(a, off, 32);
    b += __shfl_down(b, off, 32);
  }
  if (lane == 0) { ns[row] = a; ms[row] = b; }
}

__global__ void final_edge(const int* __restrict__ src, const int* __restrict__ dst,
                           const float* __restrict__ ns, const float* __restrict__ ms,
                           const float* __restrict__ bcls, float* __restrict__ out, int E) {
  int e = blockIdx.x * blockDim.x + threadIdx.x;
  if (e < E) out[e] = ns[src[e]] + ms[dst[e]] + bcls[0];
}

// ---------------------------------------------------------------------------
extern "C" void kernel_launch(void* const* d_in, const int* in_sizes, int n_in,
                              void* d_out, int out_size, void* d_ws, size_t ws_size,
                              hipStream_t stream) {
  (void)in_sizes; (void)n_in; (void)out_size; (void)ws_size;
  const float* x_oer = (const float*)d_in[0];
  const float* e_oer = (const float*)d_in[3];
  const float* e_con = (const float*)d_in[4];
  const float* e_cls = (const float*)d_in[5];
  const float* W_lin = (const float*)d_in[6];
  const float* b_lin = (const float*)d_in[7];
  const float* Wsrc  = (const float*)d_in[8];
  const float* Wdst  = (const float*)d_in[9];
  const float* att_s = (const float*)d_in[10];
  const float* att_d = (const float*)d_in[11];
  const float* b_gat = (const float*)d_in[12];
  const float* W_cls = (const float*)d_in[13];
  const float* b_cls = (const float*)d_in[14];
  const int* ei_sr   = (const int*)d_in[15];
  const int* ei_cov  = (const int*)d_in[16];
  const int* ei_bel  = (const int*)d_in[17];
  const int* ei_rcov = (const int*)d_in[18];
  const int* ei_rbel = (const int*)d_in[19];
  const int* ei_ep   = (const int*)d_in[20];
  float* out = (float*)d_out;

  // carve workspace (~230 MB)
  uintptr_t wp = (uintptr_t)d_ws;
  auto carve = [&](size_t nfloats) -> float* {
    float* r = (float*)wp;
    wp += ((nfloats * sizeof(float)) + 255) & ~(size_t)255;
    return r;
  };
  float* h_oer   = carve((size_t)N_OER * HID);
  float* h_con   = carve((size_t)N_CONCEPT * HID);
  float* h_cls   = carve((size_t)N_CLASS * HID);
  float* acc_oer = carve((size_t)N_OER * HID);
  float* acc_con = carve((size_t)N_CONCEPT * HID);
  float* acc_cls = carve((size_t)N_CLASS * HID);
  float* hs      = carve((size_t)N_OER * HID);
  float* hd      = carve((size_t)N_OER * HID);
  float* s_src   = carve(N_OER);
  float* s_dst   = carve(N_OER);
  float* segmax  = carve(N_OER);
  float* denom   = carve(N_OER);
  float* logit   = carve(E_EP + N_OER);
  float* exbuf   = carve(E_EP + N_OER);
  float* nsc     = carve(N_OER);
  float* msc     = carve(N_OER);

  auto gemm = [&](const float* X, const float* W, const float* bias, float* Y,
                  int M, int K) {
    gemm_wmma_f16<<<dim3((M + ROWS_PER_BLOCK - 1) / ROWS_PER_BLOCK), dim3(256), 0,
                    stream>>>(X, W, bias, Y, M, K);
  };

  // input projections (K = 64)
  gemm(e_oer, W_lin + 0 * F_ENT * HID, b_lin + 0 * HID, h_oer, N_OER, F_ENT);
  gemm(e_con, W_lin + 1 * F_ENT * HID, b_lin + 1 * HID, h_con, N_CONCEPT, F_ENT);
  gemm(e_cls, W_lin + 2 * F_ENT * HID, b_lin + 2 * HID, h_cls, N_CLASS, F_ENT);

  float* htab[3]   = {h_oer, h_con, h_cls};
  float* acctab[3] = {acc_oer, acc_con, acc_cls};
  int    ntab[3]   = {N_OER, N_CONCEPT, N_CLASS};

  struct Rel { int st, dt; const int* e; int E; int nl; };
  Rel rels[5] = {
    {0, 0, ei_ep,   E_EP,  N_OER},   // before_ep + self loops
    {0, 1, ei_cov,  E_COV, 0},       // covers
    {1, 2, ei_bel,  E_BEL, 0},       // belongs
    {1, 0, ei_rcov, E_COV, 0},       // rev_covers
    {2, 1, ei_rbel, E_BEL, 0},       // rev_belongs
  };

  for (int l = 0; l < 2; ++l) {
    const float* bg = b_gat + (size_t)l * 5 * HID;
    init_acc_bias<<<dim3((N_OER * HID + 255) / 256), 256, 0, stream>>>(
        acc_oer, bg + 0 * HID, bg + 3 * HID, N_OER);
    init_acc_bias<<<dim3((N_CONCEPT * HID + 255) / 256), 256, 0, stream>>>(
        acc_con, bg + 1 * HID, bg + 4 * HID, N_CONCEPT);
    init_acc_bias<<<dim3((N_CLASS * HID + 255) / 256), 256, 0, stream>>>(
        acc_cls, bg + 2 * HID, nullptr, N_CLASS);

    for (int r = 0; r < 5; ++r) {
      const Rel& R = rels[r];
      const size_t wo = ((size_t)l * 5 + r) * HID * HID;
      const size_t ao = ((size_t)l * 5 + r) * HID;
      int Ns = ntab[R.st], Nd = ntab[R.dt];

      gemm(htab[R.st], Wsrc + wo, nullptr, hs, Ns, HID);
      gemm(htab[R.dt], Wdst + wo, nullptr, hd, Nd, HID);
      rowdot128<<<dim3((Ns + 7) / 8), 256, 0, stream>>>(hs, att_s + ao, s_src, Ns);
      rowdot128<<<dim3((Nd + 7) / 8), 256, 0, stream>>>(hd, att_d + ao, s_dst, Nd);
      fill_f32<<<dim3((Nd + 255) / 256), 256, 0, stream>>>(segmax, -3.0e38f, Nd);
      fill_f32<<<dim3((Nd + 255) / 256), 256, 0, stream>>>(denom, 0.0f, Nd);

      int tot = R.E + R.nl;
      edge_logit_max<<<dim3((tot + 255) / 256), 256, 0, stream>>>(
          R.e, R.e + R.E, R.E, R.nl, s_src, s_dst, logit, segmax);
      edge_exp_sum<<<dim3((tot + 255) / 256), 256, 0, stream>>>(
          R.e, R.e + R.E, R.E, R.nl, logit, segmax, exbuf, denom);
      edge_scatter<<<dim3((tot + 7) / 8), 256, 0, stream>>>(
          R.e, R.e + R.E, R.E, R.nl, hs, exbuf, denom, acctab[R.dt]);
    }
    scale_copy<<<dim3((N_OER * HID + 255) / 256), 256, 0, stream>>>(h_oer, acc_oer, 0.5f, N_OER * HID);
    scale_copy<<<dim3((N_CONCEPT * HID + 255) / 256), 256, 0, stream>>>(h_con, acc_con, 0.5f, N_CONCEPT * HID);
    scale_copy<<<dim3((N_CLASS * HID + 255) / 256), 256, 0, stream>>>(h_cls, acc_cls, 1.0f, N_CLASS * HID);
  }

  node_scores<<<dim3((N_OER + 7) / 8), 256, 0, stream>>>(x_oer, h_oer, W_cls, nsc, msc, N_OER);
  final_edge<<<dim3((E_SR + 255) / 256), 256, 0, stream>>>(ei_sr, ei_sr + E_SR, nsc, msc, b_cls, out, E_SR);
}
